// dpnn3_79027398246425
// MI455X (gfx1250) — matrix operand order repaired and checked by reading, not compiled, not run on hardware
//
#include <hip/hip_runtime.h>
#include <hip/hip_bf16.h>

// Problem constants (dpnn3): B=8, N=2048, DK=DV=64, H=256
#define B_    8
#define N_    2048
#define DK_   64
#define H_    256
#define MTOK  (B_ * N_)   // 16384 tokens per tensor

typedef __attribute__((ext_vector_type(16))) _Float16 v16h;
typedef __attribute__((ext_vector_type(8)))  float    v8f;

union V16 { v16h v; _Float16 h[16]; uint4 u[2]; };

// Load a 16-half WMMA fragment from two 16-byte chunks.
__device__ __forceinline__ v16h ld2x16B(const _Float16* p0, const _Float16* p1) {
    V16 t;
    t.u[0] = *(const uint4*)p0;
    t.u[1] = *(const uint4*)p1;
    return t.v;
}

__device__ __forceinline__ v8f wmma_f16(v16h a, v16h b, v8f c) {
    // D = A(16x32 f16) * B(32x16 f16) + C(16x16 f32)
    return __builtin_amdgcn_wmma_f32_16x16x32_f16(false, a, false, b, (short)0, c,
                                                  false, false);
}

__device__ __forceinline__ v8f zero8() {
    v8f c;
#pragma unroll
    for (int i = 0; i < 8; ++i) c[i] = 0.0f;
    return c;
}

// ---------------------------------------------------------------------------
// Conversions
// ---------------------------------------------------------------------------
__global__ __launch_bounds__(256) void cvt_f32_f16(const float* __restrict__ in,
                                                   _Float16* __restrict__ out, int n) {
    int i = blockIdx.x * 256 + threadIdx.x;
    if (i < n) out[i] = (_Float16)in[i];
}

// W[K][Nc] row-major (fan_in x fan_out)  ->  Wt[Nc][K] row-major, f16
__global__ __launch_bounds__(256) void cvt_wt_T(const float* __restrict__ W,
                                                _Float16* __restrict__ Wt,
                                                int K, int Nc) {
    int i = blockIdx.x * 256 + threadIdx.x;
    if (i < K * Nc) {
        int n = i / K;
        int k = i - n * K;
        Wt[i] = (_Float16)W[(size_t)k * Nc + n];
    }
}

// ---------------------------------------------------------------------------
// MLP GEMM: out = act(A[M,K](f16) @ W[K,Nc] + bias), W supplied as Wt[Nc][K] f16.
// Block = 8 waves; wave w handles rows [blockIdx.x*128 + w*16, +16) x cols
// [blockIdx.y*64, +64). Compile-time K, fully unrolled, explicitly
// double-buffered so next-step loads overlap the current 4 WMMAs.
// STORET=1 writes out[b][n][i] (per-batch transposed) for the Wv branch.
// ---------------------------------------------------------------------------
template <int K, bool RELU, bool STORET>
__global__ __launch_bounds__(256) void mlp_gemm(const _Float16* __restrict__ A,
                                                const _Float16* __restrict__ Wt,
                                                const float* __restrict__ bias,
                                                _Float16* __restrict__ out,
                                                int Nc) {
    const int lane = threadIdx.x & 31;
    const int wave = threadIdx.x >> 5;
    const int lo = lane & 15, hi = lane >> 4;
    const int m0 = blockIdx.x * 128 + wave * 16;
    const int n0 = blockIdx.y * 64;

    const _Float16* Arow = A + (size_t)(m0 + lo) * K;
    const _Float16* Wrow[4];
#pragma unroll
    for (int jt = 0; jt < 4; ++jt)
        Wrow[jt] = Wt + (size_t)(n0 + jt * 16 + lo) * K;

    v8f c[4];
#pragma unroll
    for (int jt = 0; jt < 4; ++jt) c[jt] = zero8();

    constexpr int STEPS = K / 32;

    // Prologue: fragments for k-step 0.
    v16h a_cur = ld2x16B(Arow + hi * 16, Arow + hi * 16 + 8);
    v16h b_cur[4];
#pragma unroll
    for (int jt = 0; jt < 4; ++jt)
        b_cur[jt] = ld2x16B(Wrow[jt] + hi * 16, Wrow[jt] + hi * 16 + 8);

#pragma unroll
    for (int s = 0; s < STEPS; ++s) {
        v16h a_nxt;
        v16h b_nxt[4];
        if (s + 1 < STEPS) {
            const int kk = (s + 1) * 32;
            a_nxt = ld2x16B(Arow + kk + hi * 16, Arow + kk + hi * 16 + 8);
#pragma unroll
            for (int jt = 0; jt < 4; ++jt)
                b_nxt[jt] = ld2x16B(Wrow[jt] + kk + hi * 16,
                                    Wrow[jt] + kk + hi * 16 + 8);
        }
#pragma unroll
        for (int jt = 0; jt < 4; ++jt)
            c[jt] = wmma_f16(a_cur, b_cur[jt], c[jt]);
        if (s + 1 < STEPS) {
            a_cur = a_nxt;
#pragma unroll
            for (int jt = 0; jt < 4; ++jt) b_cur[jt] = b_nxt[jt];
        }
    }

#pragma unroll
    for (int jt = 0; jt < 4; ++jt) {
        const int n = n0 + jt * 16 + lo;
        const float bv = bias[n];
#pragma unroll
        for (int r = 0; r < 8; ++r) {
            const int m = m0 + r + 8 * hi;           // C layout: row = r + 8*hi
            float v = c[jt][r] + bv;
            if (RELU) v = fmaxf(v, 0.0f);
            if (STORET) {
                const int bb = m >> 11;              // m / 2048
                const int ii = m & (N_ - 1);
                out[((size_t)bb * DK_ + n) * N_ + ii] = (_Float16)v;
            } else {
                out[(size_t)m * Nc + n] = (_Float16)v;
            }
        }
    }
}

// ---------------------------------------------------------------------------
// Pass 1: per-KEY-row softmax stats over the query axis.
// dist[b,i,k] = (Wk[b,i,:] . Wq[b,k,:]) / 8 ; m_i = max_k, s_i = sum_k exp(.-m_i)
// One wave per 16 key rows; A-frags (Wk) cached in regs, two query sweeps.
// Query B-frags use two ping-pong sets (unroll-2) so no register copies:
// each set is reloaded for tile t+2 right after its WMMAs consume it.
// ---------------------------------------------------------------------------
__global__ __launch_bounds__(256) void attn_stats(const _Float16* __restrict__ Wk,
                                                  const _Float16* __restrict__ Wq,
                                                  float* __restrict__ m_out,
                                                  float* __restrict__ s_out) {
    const int lane = threadIdx.x & 31;
    const int wave = threadIdx.x >> 5;
    const int lo = lane & 15, hi = lane >> 4;
    const int g  = blockIdx.x * 8 + wave;   // 0 .. B*N/16-1
    const int b  = g >> 7;                  // 128 groups per batch
    const int i0 = (g & 127) * 16;

    const _Float16* krow = Wk + ((size_t)b * N_ + i0 + lo) * DK_;
    const v16h a0 = ld2x16B(krow + hi * 16,      krow + hi * 16 + 8);
    const v16h a1 = ld2x16B(krow + 32 + hi * 16, krow + 32 + hi * 16 + 8);
    const _Float16* qbase = Wq + (size_t)b * N_ * DK_ + (size_t)lo * DK_ + hi * 16;

    float mx[8];
#pragma unroll
    for (int r = 0; r < 8; ++r) mx[r] = -3.402823466e38f;

    // Sweep 1: per-lane row maxima (lane sees 1 of 16 columns per tile)
    {
        v16h s0a = ld2x16B(qbase, qbase + 8);
        v16h s0b = ld2x16B(qbase + 32, qbase + 40);
        const _Float16* q1 = qbase + (size_t)16 * DK_;
        v16h s1a = ld2x16B(q1, q1 + 8);
        v16h s1b = ld2x16B(q1 + 32, q1 + 40);
        for (int q0 = 0; q0 < N_; q0 += 32) {
            v8f c = zero8();
            c = wmma_f16(a0, s0a, c);
            c = wmma_f16(a1, s0b, c);
            if (q0 + 32 < N_) {
                const _Float16* qn = qbase + (size_t)(q0 + 32) * DK_;
                s0a = ld2x16B(qn, qn + 8);
                s0b = ld2x16B(qn + 32, qn + 40);
            }
#pragma unroll
            for (int r = 0; r < 8; ++r) mx[r] = fmaxf(mx[r], c[r]);

            v8f c2 = zero8();
            c2 = wmma_f16(a0, s1a, c2);
            c2 = wmma_f16(a1, s1b, c2);
            if (q0 + 48 < N_) {
                const _Float16* qn = qbase + (size_t)(q0 + 48) * DK_;
                s1a = ld2x16B(qn, qn + 8);
                s1b = ld2x16B(qn + 32, qn + 40);
            }
#pragma unroll
            for (int r = 0; r < 8; ++r) mx[r] = fmaxf(mx[r], c2[r]);
        }
    }
    // Butterfly over the 16 lanes of each half -> true row max, then scale
#pragma unroll
    for (int r = 0; r < 8; ++r) {
        for (int d = 1; d < 16; d <<= 1) mx[r] = fmaxf(mx[r], __shfl_xor(mx[r], d));
        mx[r] *= 0.125f;   // 1/sqrt(64)
    }

    float sm[8];
#pragma unroll
    for (int r = 0; r < 8; ++r) sm[r] = 0.0f;

    // Sweep 2: exp-sums with known max (same ping-pong structure)
    {
        v16h s0a = ld2x16B(qbase, qbase + 8);
        v16h s0b = ld2x16B(qbase + 32, qbase + 40);
        const _Float16* q1 = qbase + (size_t)16 * DK_;
        v16h s1a = ld2x16B(q1, q1 + 8);
        v16h s1b = ld2x16B(q1 + 32, q1 + 40);
        for (int q0 = 0; q0 < N_; q0 += 32) {
            v8f c = zero8();
            c = wmma_f16(a0, s0a, c);
            c = wmma_f16(a1, s0b, c);
            if (q0 + 32 < N_) {
                const _Float16* qn = qbase + (size_t)(q0 + 32) * DK_;
                s0a = ld2x16B(qn, qn + 8);
                s0b = ld2x16B(qn + 32, qn + 40);
            }
#pragma unroll
            for (int r = 0; r < 8; ++r) sm[r] += __expf(c[r] * 0.125f - mx[r]);

            v8f c2 = zero8();
            c2 = wmma_f16(a0, s1a, c2);
            c2 = wmma_f16(a1, s1b, c2);
            if (q0 + 48 < N_) {
                const _Float16* qn = qbase + (size_t)(q0 + 48) * DK_;
                s1a = ld2x16B(qn, qn + 8);
                s1b = ld2x16B(qn + 32, qn + 40);
            }
#pragma unroll
            for (int r = 0; r < 8; ++r) sm[r] += __expf(c2[r] * 0.125f - mx[r]);
        }
    }
#pragma unroll
    for (int r = 0; r < 8; ++r)
        for (int d = 1; d < 16; d <<= 1) sm[r] += __shfl_xor(sm[r], d);

    if (lo == 0) {
#pragma unroll
        for (int r = 0; r < 8; ++r) {
            const size_t row = (size_t)b * N_ + i0 + r + 8 * hi;
            m_out[row] = mx[r];
            s_out[row] = sm[r];
        }
    }
}

// ---------------------------------------------------------------------------
// Pass 2: context[b,q,j] = sum_i P[i,q] * Wv[i,j].
// Wave = 16 queries x all 64 j. Per 32-key half-step: recompute two 16x16
// dist tiles (4 WMMA), exponentiate with LDS-staged stats. The f32
// C-fragment of dist maps *exactly* onto the f16 A-fragment of the
// transposed P tile (no shuffles), then 4 PV WMMAs against Wv^T.
// Key A-fragments use two ping-pong sets (unroll-2 over 64 keys): each set
// is reloaded for i0+64 right after its dist WMMAs, so the exp phase and the
// other half's compute cover the load latency — no register copies.
// ---------------------------------------------------------------------------
__global__ __launch_bounds__(256) void attn_ctx(const _Float16* __restrict__ Wk,
                                                const _Float16* __restrict__ Wq,
                                                const _Float16* __restrict__ Wvt,
                                                const float* __restrict__ m_in,
                                                const float* __restrict__ s_in,
                                                float* __restrict__ out) {
    __shared__ float sm[N_];
    __shared__ float srs[N_];

    const int lane = threadIdx.x & 31;
    const int wave = threadIdx.x >> 5;
    const int lo = lane & 15, hi = lane >> 4;
    const int g  = blockIdx.x * 8 + wave;
    const int b  = g >> 7;                 // all 8 waves of a block share b
    const int q0 = (g & 127) * 16;

    for (int idx = threadIdx.x; idx < N_; idx += 256) {
        sm[idx]  = m_in[(size_t)b * N_ + idx];
        srs[idx] = 1.0f / s_in[(size_t)b * N_ + idx];
    }
    __syncthreads();

    const _Float16* qrow = Wq + ((size_t)b * N_ + q0 + lo) * DK_ + hi * 8;
    const v16h bq0 = ld2x16B(qrow, qrow + 16);
    const v16h bq1 = ld2x16B(qrow + 32, qrow + 48);

    const _Float16* kbase = Wk  + ((size_t)b * N_ + lo) * DK_;
    const _Float16* vbase = Wvt + (size_t)b * DK_ * N_;

    v8f ctx[4];
#pragma unroll
    for (int jt = 0; jt < 4; ++jt) ctx[jt] = zero8();

    // Prologue: A-fragment sets for key blocks 0 (set 0) and 32 (set 1).
    v16h aS[2][4];
#pragma unroll
    for (int half = 0; half < 2; ++half) {
        const _Float16* kr0 = kbase + (size_t)(half * 32) * DK_;
        const _Float16* kr1 = kr0 + 16 * DK_;
        aS[half][0] = ld2x16B(kr0 + hi * 8,      kr0 + 16 + hi * 8);
        aS[half][1] = ld2x16B(kr0 + 32 + hi * 8, kr0 + 48 + hi * 8);
        aS[half][2] = ld2x16B(kr1 + hi * 8,      kr1 + 16 + hi * 8);
        aS[half][3] = ld2x16B(kr1 + 32 + hi * 8, kr1 + 48 + hi * 8);
    }

    for (int ib = 0; ib < N_; ib += 64) {
#pragma unroll
        for (int half = 0; half < 2; ++half) {
            const int i0 = ib + half * 32;

            // Issue PV operand loads early (consumed after the exp phase).
            v16h bv[4];
#pragma unroll
            for (int jt = 0; jt < 4; ++jt) {
                const _Float16* vr = vbase + (size_t)(jt * 16 + lo) * N_ + i0 + hi * 8;
                bv[jt] = ld2x16B(vr, vr + 16);
            }

            v8f c0 = zero8();
            c0 = wmma_f16(aS[half][0], bq0, c0);
            c0 = wmma_f16(aS[half][1], bq1, c0);
            v8f c1 = zero8();
            c1 = wmma_f16(aS[half][2], bq0, c1);
            c1 = wmma_f16(aS[half][3], bq1, c1);

            // Reload this set for key block i0+64 (same parity, consumed one
            // full outer iteration later).
            if (i0 + 64 < N_) {
                const _Float16* kr0 = kbase + (size_t)(i0 + 64) * DK_;
                const _Float16* kr1 = kr0 + 16 * DK_;
                __builtin_prefetch(kr0 + 64 * DK_, 0, 1);   // -> global_prefetch
                aS[half][0] = ld2x16B(kr0 + hi * 8,      kr0 + 16 + hi * 8);
                aS[half][1] = ld2x16B(kr0 + 32 + hi * 8, kr0 + 48 + hi * 8);
                aS[half][2] = ld2x16B(kr1 + hi * 8,      kr1 + 16 + hi * 8);
                aS[half][3] = ld2x16B(kr1 + 32 + hi * 8, kr1 + 48 + hi * 8);
            }

            // Stats for keys i0+8*hi+r (c0) and i0+16+8*hi+r (c1)
            float mv0[8], rv0[8], mv1[8], rv1[8];
            *(float4*)&mv0[0] = *(const float4*)&sm [i0 + hi * 8];
            *(float4*)&mv0[4] = *(const float4*)&sm [i0 + hi * 8 + 4];
            *(float4*)&rv0[0] = *(const float4*)&srs[i0 + hi * 8];
            *(float4*)&rv0[4] = *(const float4*)&srs[i0 + hi * 8 + 4];
            *(float4*)&mv1[0] = *(const float4*)&sm [i0 + 16 + hi * 8];
            *(float4*)&mv1[4] = *(const float4*)&sm [i0 + 16 + hi * 8 + 4];
            *(float4*)&rv1[0] = *(const float4*)&srs[i0 + 16 + hi * 8];
            *(float4*)&rv1[4] = *(const float4*)&srs[i0 + 16 + hi * 8 + 4];

            // C-fragment (16 keys x 16 queries) -> A-fragment of P^T
            // (16 q x 32 keys): lane lo is query row; elems 0-7 = keys
            // 8*hi..(c0), elems 8-15 = keys 16+8*hi..(c1). Exact layout
            // match, no cross-lane traffic.
            V16 af;
#pragma unroll
            for (int r = 0; r < 8; ++r) {
                af.h[r]     = (_Float16)(__expf(c0[r] * 0.125f - mv0[r]) * rv0[r]);
                af.h[r + 8] = (_Float16)(__expf(c1[r] * 0.125f - mv1[r]) * rv1[r]);
            }

#pragma unroll
            for (int jt = 0; jt < 4; ++jt)
                ctx[jt] = wmma_f16(af.v, bv[jt], ctx[jt]);
        }
    }

#pragma unroll
    for (int jt = 0; jt < 4; ++jt) {
        const int j = jt * 16 + lo;
#pragma unroll
        for (int r = 0; r < 8; ++r) {
            const int q = q0 + r + 8 * hi;
            out[((size_t)b * N_ + q) * DK_ + j] = ctx[jt][r];
        }
    }
}

// ---------------------------------------------------------------------------
// Host driver
// ---------------------------------------------------------------------------
extern "C" void kernel_launch(void* const* d_in, const int* in_sizes, int n_in,
                              void* d_out, int out_size, void* d_ws, size_t ws_size,
                              hipStream_t stream) {
    (void)in_sizes; (void)n_in; (void)out_size; (void)ws_size;

    const float* KEY   = (const float*)d_in[0];
    const float* VALUE = (const float*)d_in[1];
    const float* QUERY = (const float*)d_in[2];
    const float* Wk1 = (const float*)d_in[3];  const float* bk1 = (const float*)d_in[4];
    const float* Wk2 = (const float*)d_in[5];  const float* bk2 = (const float*)d_in[6];
    const float* Wk3 = (const float*)d_in[7];  const float* bk3 = (const float*)d_in[8];
    const float* Wv1 = (const float*)d_in[9];  const float* bv1 = (const float*)d_in[10];
    const float* Wv2 = (const float*)d_in[11]; const float* bv2 = (const float*)d_in[12];
    const float* Wv3 = (const float*)d_in[13]; const float* bv3 = (const float*)d_in[14];

    char* ws = (char*)d_ws;
    size_t off = 0;
    auto alloc = [&](size_t bytes) -> char* {
        char* p = ws + off;
        off += (bytes + 255) & ~(size_t)255;
        return p;
    };

    _Float16* aAct  = (_Float16*)alloc((size_t)MTOK * DK_ * 2);   // 2 MB (reused K/Q/V)
    _Float16* h1    = (_Float16*)alloc((size_t)MTOK * H_  * 2);   // 8 MB
    _Float16* h2    = (_Float16*)alloc((size_t)MTOK * H_  * 2);   // 8 MB
    _Float16* Wk16  = (_Float16*)alloc((size_t)MTOK * DK_ * 2);
    _Float16* Wq16  = (_Float16*)alloc((size_t)MTOK * DK_ * 2);
    _Float16* Wvt16 = (_Float16*)alloc((size_t)MTOK * DK_ * 2);   // [b][64][2048]
    _Float16* W1t   = (_Float16*)alloc((size_t)DK_ * H_ * 2);
    _Float16* W2t   = (_Float16*)alloc((size_t)H_  * H_ * 2);
    _Float16* W3t   = (_Float16*)alloc((size_t)H_  * DK_ * 2);
    float*    m_arr = (float*)alloc((size_t)MTOK * 4);
    float*    s_arr = (float*)alloc((size_t)MTOK * 4);

    auto cvt = [&](const float* src, _Float16* dst, int n) {
        cvt_f32_f16<<<dim3((n + 255) / 256), dim3(256), 0, stream>>>(src, dst, n);
    };
    auto cvtw = [&](const float* W, _Float16* Wt, int K, int Nc) {
        int n = K * Nc;
        cvt_wt_T<<<dim3((n + 255) / 256), dim3(256), 0, stream>>>(W, Wt, K, Nc);
    };

    const dim3 gemmBlk(256);
    const dim3 grid64x256(MTOK / 128, H_ / 64);   // Nc = 256
    const dim3 gridNc64(MTOK / 128, 1);           // Nc = 64

    // K/Q branch (shared weights)
    cvtw(Wk1, W1t, DK_, H_);
    cvtw(Wk2, W2t, H_, H_);
    cvtw(Wk3, W3t, H_, DK_);

    cvt(KEY, aAct, MTOK * DK_);
    mlp_gemm<DK_, true,  false><<<grid64x256, gemmBlk, 0, stream>>>(aAct, W1t, bk1, h1, H_);
    mlp_gemm<H_,  true,  false><<<grid64x256, gemmBlk, 0, stream>>>(h1,   W2t, bk2, h2, H_);
    mlp_gemm<H_,  false, false><<<gridNc64,   gemmBlk, 0, stream>>>(h2,   W3t, bk3, Wk16, DK_);

    cvt(QUERY, aAct, MTOK * DK_);
    mlp_gemm<DK_, true,  false><<<grid64x256, gemmBlk, 0, stream>>>(aAct, W1t, bk1, h1, H_);
    mlp_gemm<H_,  true,  false><<<grid64x256, gemmBlk, 0, stream>>>(h1,   W2t, bk2, h2, H_);
    mlp_gemm<H_,  false, false><<<gridNc64,   gemmBlk, 0, stream>>>(h2,   W3t, bk3, Wq16, DK_);

    // V branch (weights reuse the same transposed buffers)
    cvtw(Wv1, W1t, DK_, H_);
    cvtw(Wv2, W2t, H_, H_);
    cvtw(Wv3, W3t, H_, DK_);

    cvt(VALUE, aAct, MTOK * DK_);
    mlp_gemm<DK_, true,  false><<<grid64x256, gemmBlk, 0, stream>>>(aAct, W1t, bv1, h1, H_);
    mlp_gemm<H_,  true,  false><<<grid64x256, gemmBlk, 0, stream>>>(h1,   W2t, bv2, h2, H_);
    mlp_gemm<H_,  false, true ><<<gridNc64,   gemmBlk, 0, stream>>>(h2,   W3t, bv3, Wvt16, DK_);

    // Attention
    attn_stats<<<dim3(128), dim3(256), 0, stream>>>(Wk16, Wq16, m_arr, s_arr);
    attn_ctx  <<<dim3(128), dim3(256), 0, stream>>>(Wk16, Wq16, Wvt16, m_arr, s_arr,
                                                    (float*)d_out);
}
